// MultiHeadAttention_88905823027417
// MI455X (gfx1250) — matrix operand order repaired and checked by reading, not compiled, not run on hardware
//
#include <hip/hip_runtime.h>

// ---------------- CDNA5 WMMA types ----------------
typedef __attribute__((ext_vector_type(16))) __bf16 v16bf;
typedef __attribute__((ext_vector_type(8)))  __bf16 v8bf;
typedef __attribute__((ext_vector_type(8)))  float  v8f;

union ABfrag { v16bf v; v8bf h[2]; __bf16 e[16]; };
union F8     { v8f  v; float f[8]; };

static __device__ __forceinline__ v8f wmma_bf16(v16bf a, v16bf b, v8f c) {
    // D(16x16,f32) = A(16x32,bf16) x B(32x16,bf16) + C
    return __builtin_amdgcn_wmma_f32_16x16x32_bf16(false, a, false, b, (short)0, c, false, false);
}

// Problem dims
#define BB 4
#define SS 2048
#define DD 1024
#define HH 16
#define HD 64
#define MROWS (BB*SS)          // 8192

// ---------------- kernel 0a: f32 -> bf16 convert ----------------
__global__ void f2bf_kernel(const float* __restrict__ src, __bf16* __restrict__ dst, int n) {
    for (int i = blockIdx.x * blockDim.x + threadIdx.x; i < n; i += gridDim.x * blockDim.x)
        dst[i] = (__bf16)src[i];
}

// ---------------- kernel 0b: RoPE cos/sin tables [S,32] ----------------
__global__ void rope_table_kernel(float* __restrict__ cosT, float* __restrict__ sinT) {
    int t = blockIdx.x * blockDim.x + threadIdx.x;     // 2048*32 threads
    int s = t >> 5, idx = t & 31;
    float inv = __expf(-(float)idx * (9.210340372f / 32.0f));  // 10000^(-idx/32)
    float ang = (float)s * inv;
    float sn, cs;
    sincosf(ang, &sn, &cs);
    cosT[t] = cs;
    sinT[t] = sn;
}

// ---------------- kernel 1: QKV GEMM (+RoPE) ----------------
// Each wave: 32(M) x 64(N) tile of [8192 x 3072] = xb @ wqkvb.
// 64-col chunk == one (section, head) pair. chunk 0..15 = Q heads, 16..31 = K heads, 32..47 = V heads.
__global__ void qkv_rope_kernel(const __bf16* __restrict__ xb,     // [8192,1024]
                                const __bf16* __restrict__ wb,     // [1024,3072]
                                const float*  __restrict__ cosT,   // [2048,32]
                                const float*  __restrict__ sinT,   // [2048,32]
                                __bf16* __restrict__ Qb,           // [B,H,S,64]
                                __bf16* __restrict__ Ktb,          // [B,H,64,S]  (transposed K)
                                __bf16* __restrict__ Vb)           // [B,H,S,64]
{
    const int lane  = threadIdx.x & 31;
    const int wid   = blockIdx.x * (blockDim.x >> 5) + (threadIdx.x >> 5);
    const int chunk = wid % 48;          // which 64-wide column chunk of 3072
    const int mtile = wid / 48;          // which 32-row tile of 8192
    const int c0 = chunk * 64;
    const int m0 = mtile * 32;

    const int mfrag = lane & 15;         // A-frag row
    const int kb    = (lane >> 4) * 8;   // A-frag K base
    const int half  = lane >> 4;
    const int n     = lane & 15;

    v8f acc[2][4] = {};
    for (int k0 = 0; k0 < DD; k0 += 32) {
        ABfrag a[2];
        #pragma unroll
        for (int sub = 0; sub < 2; ++sub) {
            const __bf16* ap = xb + (size_t)(m0 + sub * 16 + mfrag) * DD + k0 + kb;
            a[sub].h[0] = *(const v8bf*)(ap);
            a[sub].h[1] = *(const v8bf*)(ap + 16);
        }
        #pragma unroll
        for (int j = 0; j < 4; ++j) {
            ABfrag bfu;
            #pragma unroll
            for (int s = 0; s < 16; ++s) bfu.e[s] = wb[(size_t)(k0 + kb + (s & 7) + ((s >> 3) << 4)) * 3072 + c0 + j * 16 + n];
            v16bf bf = bfu.v;
            acc[0][j] = wmma_bf16(a[0].v, bf, acc[0][j]);
            acc[1][j] = wmma_bf16(a[1].v, bf, acc[1][j]);
        }
    }

    const int sect = c0 / DD;            // 0=Q 1=K 2=V
    const int h    = (c0 % DD) / HD;

    #pragma unroll
    for (int sub = 0; sub < 2; ++sub) {
        F8 A[4];
        #pragma unroll
        for (int j = 0; j < 4; ++j) A[j].v = acc[sub][j];

        #pragma unroll
        for (int i = 0; i < 8; ++i) {
            int row = m0 + sub * 16 + i + half * 8;    // global (b,s) row
            int b = row >> 11;
            int s = row & (SS - 1);

            float outv[4];
            if (sect < 2) {
                #pragma unroll
                for (int jj = 0; jj < 2; ++jj) {
                    int idx = jj * 16 + n;                    // d%32 for both paired chunks
                    float cs = cosT[s * 32 + idx];
                    float sn = sinT[s * 32 + idx];
                    float a_lo = A[jj].f[i];                  // d = idx
                    float a_hi = A[jj + 2].f[i];              // d = idx + 32
                    outv[jj]     = a_lo * cs - a_hi * sn;     // rotate_half: -x2 part
                    outv[jj + 2] = a_hi * cs + a_lo * sn;     //              +x1 part
                }
            } else {
                #pragma unroll
                for (int j = 0; j < 4; ++j) outv[j] = A[j].f[i];
            }

            #pragma unroll
            for (int j = 0; j < 4; ++j) {
                int d = j * 16 + n;
                __bf16 val = (__bf16)outv[j];
                if (sect == 0) {
                    Qb[((size_t)(b * HH + h) * SS + s) * HD + d] = val;
                } else if (sect == 1) {
                    Ktb[((size_t)(b * HH + h) * HD + d) * SS + s] = val;
                } else {
                    Vb[((size_t)(b * HH + h) * SS + s) * HD + d] = val;
                }
            }
        }
    }
}

// ---------------- kernel 2: flash attention ----------------
// One wave per 16-query tile of one (b,h). Online softmax over 32-key chunks.
// 8 consecutive waves in a block cover consecutive q-tiles of the same (b,h),
// so K/V chunk loads hit the WGP caches.
__global__ void attn_kernel(const __bf16* __restrict__ Qb,
                            const __bf16* __restrict__ Ktb,
                            const __bf16* __restrict__ Vb,
                            const int*    __restrict__ amask,   // [B,S]
                            __bf16*       __restrict__ ctxb)    // [B,S,H*64]
{
    __shared__ __bf16 lds[8][16 * 32];   // per-wave P tile (16 q x 32 k)

    const int lane   = threadIdx.x & 31;
    const int wlocal = threadIdx.x >> 5;
    const int wid    = blockIdx.x * (blockDim.x >> 5) + wlocal;
    const int qtile  = wid & (SS / 16 - 1);   // 0..127
    const int bh     = wid >> 7;              // 0..63
    const int b = bh >> 4, h = bh & 15;
    const int q0 = qtile * 16;

    const __bf16* Q  = Qb  + (size_t)bh * SS * HD;
    const __bf16* Kt = Ktb + (size_t)bh * HD * SS;
    const __bf16* V  = Vb  + (size_t)bh * SS * HD;

    const int mfrag = lane & 15;
    const int kb    = (lane >> 4) * 8;
    const int half  = lane >> 4;
    const int n     = lane & 15;

    // Q A-fragments for d-chunks [0,32) and [32,64)
    ABfrag qf[2];
    #pragma unroll
    for (int dd = 0; dd < 2; ++dd) {
        const __bf16* qp = Q + (size_t)(q0 + mfrag) * HD + dd * 32 + kb;
        qf[dd].h[0] = *(const v8bf*)(qp);
        qf[dd].h[1] = *(const v8bf*)(qp + 16);
    }

    float mrow[8], lrow[8];
    F8 o[4] = {};
    #pragma unroll
    for (int i = 0; i < 8; ++i) { mrow[i] = -3.0e38f; lrow[i] = 0.0f; }

    const float scale = 0.125f;          // 1/sqrt(64)
    __bf16* pl = &lds[wlocal][0];

    for (int kc = 0; kc <= q0 + 15; kc += 32) {
        // ---- scores: 16 q x 32 k ----
        v8f sa = {}, sb = {};
        #pragma unroll
        for (int dd = 0; dd < 2; ++dd) {
            ABfrag bku0, bku1;
            #pragma unroll
            for (int s = 0; s < 16; ++s) {
                size_t kr = (size_t)(dd * 32 + kb + (s & 7) + ((s >> 3) << 4)) * SS;
                bku0.e[s] = Kt[kr + kc + n];
                bku1.e[s] = Kt[kr + kc + 16 + n];
            }
            v16bf bk0 = bku0.v;
            v16bf bk1 = bku1.v;
            sa = wmma_bf16(qf[dd].v, bk0, sa);
            sb = wmma_bf16(qf[dd].v, bk1, sb);
        }
        F8 S0, S1; S0.v = sa; S1.v = sb;

        const int col0 = kc + n, col1 = kc + 16 + n;
        const int pad0 = amask[b * SS + col0];
        const int pad1 = amask[b * SS + col1];

        // ---- online softmax per row ----
        #pragma unroll
        for (int i = 0; i < 8; ++i) {
            int row = q0 + i + half * 8;
            float v0 = S0.f[i] * scale; if (col0 > row || pad0 == 0) v0 = -3.0e38f;
            float v1 = S1.f[i] * scale; if (col1 > row || pad1 == 0) v1 = -3.0e38f;

            float mx = fmaxf(v0, v1);
            #pragma unroll
            for (int off = 1; off <= 8; off <<= 1) mx = fmaxf(mx, __shfl_xor(mx, off, 32));
            float mn  = fmaxf(mrow[i], mx);
            float fac = __expf(mrow[i] - mn);
            float p0  = __expf(v0 - mn);
            float p1  = __expf(v1 - mn);
            float ps  = p0 + p1;
            #pragma unroll
            for (int off = 1; off <= 8; off <<= 1) ps += __shfl_xor(ps, off, 32);
            lrow[i] = lrow[i] * fac + ps;
            mrow[i] = mn;
            #pragma unroll
            for (int dd = 0; dd < 4; ++dd) o[dd].f[i] *= fac;

            int mloc = i + half * 8;                 // local row of this wave's P tile
            pl[mloc * 32 + n]      = (__bf16)p0;
            pl[mloc * 32 + 16 + n] = (__bf16)p1;
        }

        // ---- reshape P (C-layout -> A-layout) through LDS ----
        asm volatile("s_wait_dscnt 0" ::: "memory");
        ABfrag pf;
        const __bf16* pp = pl + mfrag * 32 + kb;
        pf.h[0] = *(const v8bf*)(pp);
        pf.h[1] = *(const v8bf*)(pp + 16);

        // ---- O += P x V ----
        #pragma unroll
        for (int dd = 0; dd < 4; ++dd) {
            ABfrag bvu;
            #pragma unroll
            for (int s = 0; s < 16; ++s) bvu.e[s] = V[(size_t)(kc + kb + (s & 7) + ((s >> 3) << 4)) * HD + dd * 16 + n];
            v16bf bv = bvu.v;
            o[dd].v = wmma_bf16(pf.v, bv, o[dd].v);
        }
    }

    // ---- normalize, write context in [b,s,h*64+d] layout ----
    #pragma unroll
    for (int i = 0; i < 8; ++i) {
        float inv = 1.0f / lrow[i];
        int s = q0 + i + half * 8;
        #pragma unroll
        for (int dd = 0; dd < 4; ++dd) {
            int d = dd * 16 + n;
            ctxb[((size_t)(b * SS + s)) * DD + h * HD + d] = (__bf16)(o[dd].f[i] * inv);
        }
    }
}

// ---------------- kernel 3: output projection ----------------
// Each wave: 32(M) x 64(N) tile of [8192 x 1024] = ctxb @ woutb, f32 out.
__global__ void out_proj_kernel(const __bf16* __restrict__ ctxb,  // [8192,1024]
                                const __bf16* __restrict__ wob,   // [1024,1024]
                                float* __restrict__ out)          // [8192,1024]
{
    const int lane  = threadIdx.x & 31;
    const int wid   = blockIdx.x * (blockDim.x >> 5) + (threadIdx.x >> 5);
    const int chunk = wid % 16;
    const int mtile = wid / 16;
    const int c0 = chunk * 64;
    const int m0 = mtile * 32;

    const int mfrag = lane & 15;
    const int kb    = (lane >> 4) * 8;
    const int half  = lane >> 4;
    const int n     = lane & 15;

    v8f acc[2][4] = {};
    for (int k0 = 0; k0 < DD; k0 += 32) {
        ABfrag a[2];
        #pragma unroll
        for (int sub = 0; sub < 2; ++sub) {
            const __bf16* ap = ctxb + (size_t)(m0 + sub * 16 + mfrag) * DD + k0 + kb;
            a[sub].h[0] = *(const v8bf*)(ap);
            a[sub].h[1] = *(const v8bf*)(ap + 16);
        }
        #pragma unroll
        for (int j = 0; j < 4; ++j) {
            ABfrag bfu;
            #pragma unroll
            for (int s = 0; s < 16; ++s) bfu.e[s] = wob[(size_t)(k0 + kb + (s & 7) + ((s >> 3) << 4)) * DD + c0 + j * 16 + n];
            v16bf bf = bfu.v;
            acc[0][j] = wmma_bf16(a[0].v, bf, acc[0][j]);
            acc[1][j] = wmma_bf16(a[1].v, bf, acc[1][j]);
        }
    }

    #pragma unroll
    for (int sub = 0; sub < 2; ++sub) {
        #pragma unroll
        for (int j = 0; j < 4; ++j) {
            F8 r; r.v = acc[sub][j];
            #pragma unroll
            for (int i = 0; i < 8; ++i) {
                int row = m0 + sub * 16 + i + half * 8;
                out[(size_t)row * DD + c0 + j * 16 + n] = r.f[i];
            }
        }
    }
}

// ---------------- host launcher ----------------
extern "C" void kernel_launch(void* const* d_in, const int* in_sizes, int n_in,
                              void* d_out, int out_size, void* d_ws, size_t ws_size,
                              hipStream_t stream) {
    const float* x     = (const float*)d_in[0];   // [4,2048,1024]
    const int*   amask = (const int*)  d_in[1];   // [4,2048]
    const float* wqkv  = (const float*)d_in[2];   // [1024,3072]
    const float* wout  = (const float*)d_in[3];   // [1024,1024]
    float* out = (float*)d_out;

    char* ws = (char*)d_ws;
    size_t off = 0;
    __bf16* xb    = (__bf16*)(ws + off); off += (size_t)MROWS * DD * 2;       // 16 MB
    __bf16* wqb   = (__bf16*)(ws + off); off += (size_t)DD * 3 * DD * 2;      // 6 MB
    __bf16* wob   = (__bf16*)(ws + off); off += (size_t)DD * DD * 2;          // 2 MB
    __bf16* Qb    = (__bf16*)(ws + off); off += (size_t)BB * HH * SS * HD * 2;
    __bf16* Ktb   = (__bf16*)(ws + off); off += (size_t)BB * HH * HD * SS * 2;
    __bf16* Vb    = (__bf16*)(ws + off); off += (size_t)BB * HH * SS * HD * 2;
    __bf16* ctxb  = (__bf16*)(ws + off); off += (size_t)MROWS * DD * 2;
    float*  cosT  = (float*)(ws + off);  off += (size_t)SS * 32 * 4;          // 256 KB
    float*  sinT  = (float*)(ws + off);  off += (size_t)SS * 32 * 4;          // 256 KB

    // 0) precision conversion + RoPE tables
    f2bf_kernel<<<2048, 256, 0, stream>>>(x,    xb,  MROWS * DD);
    f2bf_kernel<<<2048, 256, 0, stream>>>(wqkv, wqb, DD * 3 * DD);
    f2bf_kernel<<<1024, 256, 0, stream>>>(wout, wob, DD * DD);
    rope_table_kernel<<<SS * 32 / 256, 256, 0, stream>>>(cosT, sinT);

    // 1) QKV projection + RoPE (+K transpose). 256 m-tiles x 48 chunks = 12288 waves / 8 per block
    qkv_rope_kernel<<<1536, 256, 0, stream>>>(xb, wqb, cosT, sinT, Qb, Ktb, Vb);

    // 2) flash attention. 64 (b,h) x 128 q-tiles = 8192 waves / 8 per block
    attn_kernel<<<1024, 256, 0, stream>>>(Qb, Ktb, Vb, amask, ctxb);

    // 3) output projection. 256 m-tiles x 16 chunks = 4096 waves / 8 per block
    out_proj_kernel<<<512, 256, 0, stream>>>(ctxb, wob, out);
}
